// LSTMModel_46136538694211
// MI455X (gfx1250) — matrix operand order repaired and checked by reading, not compiled, not run on hardware
//
#include <hip/hip_runtime.h>

// Problem dims (fixed by the reference)
#define BB 64
#define TT 512
#define DD 128
#define HH 512
#define ULDS_STRIDE 520   // 512 + 8 bf16 pad -> lanes 4 dword-banks apart (conflict-free)

typedef __attribute__((ext_vector_type(16))) __bf16 v16bf;
typedef __attribute__((ext_vector_type(8)))  float  v8f;

struct B32x8 { uint4 lo, hi; };

// --- helpers -------------------------------------------------------------

__device__ inline __bf16 f2bf(float f) {
  unsigned u = __builtin_bit_cast(unsigned, f);
  unsigned r = (u + 0x7FFFu + ((u >> 16) & 1u)) >> 16;   // RNE
  return __builtin_bit_cast(__bf16, (unsigned short)r);
}

// A-fragment (16-bit A 16x32): lane<16 -> row, K = {koff..koff+15} contiguous
__device__ inline v16bf load_frag_a(const __bf16* p) {
  B32x8 r;
  r.lo = *(const uint4*)(p);
  r.hi = *(const uint4*)(p + 8);
  return __builtin_bit_cast(v16bf, r);
}

// B-fragment: column n = row of (row-major) weight, 16 contiguous bf16
__device__ inline v16bf load_frag_b(const __bf16* p) {
  B32x8 r;
  r.lo = *(const uint4*)(p);
  r.hi = *(const uint4*)(p + 8);
  return __builtin_bit_cast(v16bf, r);
}

__device__ inline v8f wmma_bf16(v16bf a, v16bf b, v8f c) {
  return __builtin_amdgcn_wmma_f32_16x16x32_bf16(
      /*neg_a=*/false, a, /*neg_b=*/false, b,
      /*c_mod=*/(short)0, c, /*reuse_a=*/false, /*reuse_b=*/false);
}

__device__ inline float fsigmoid(float x) { return 1.0f / (1.0f + __expf(-x)); }
__device__ inline float ftanh(float x)    { return 2.0f / (1.0f + __expf(-2.0f * x)) - 1.0f; }

// --- conversion / init ---------------------------------------------------

__global__ void cvt_bf16_kernel(__bf16* __restrict__ dst,
                                const float* __restrict__ src, int n) {
  int i = blockIdx.x * blockDim.x + threadIdx.x;
  if (i < n) dst[i] = f2bf(src[i]);
}

__global__ void init_kernel(__bf16* __restrict__ hbuf, unsigned* __restrict__ syn,
                            float* __restrict__ ys, const float* __restrict__ out_b) {
  int i = blockIdx.x * blockDim.x + threadIdx.x;
  if (i < 2 * BB * HH) hbuf[i] = f2bf(0.0f);
  if (i < 2) syn[i] = 0u;
  if (i < BB * TT) ys[i] = out_b[0];
}

// --- phase 1: gate pre-activation GEMM ----------------------------------
// X[bf16, M=B*T x K=128] * W^T (4 gates) -> gates[t][gate][b][h] (f32, +bias)

__global__ __launch_bounds__(256) void gate_gemm_kernel(
    const __bf16* __restrict__ xbf, const __bf16* __restrict__ Wbf,
    const float* __restrict__ bi, const float* __restrict__ bf_,
    const float* __restrict__ bo, const float* __restrict__ bg,
    float* __restrict__ gates)
{
  const int lane = threadIdx.x & 31;
  const int wid  = blockIdx.x * (blockDim.x >> 5) + (threadIdx.x >> 5);
  const int m0 = (wid >> 5) << 4;   // M = 32768
  const int h0 = (wid & 31) << 4;   // H = 512
  const int nl = lane & 15;
  const int half = lane >> 4;

  const __bf16* arow = xbf + (m0 + nl) * DD + (half << 4);
  const __bf16* w0 = Wbf + (0 * HH + h0 + nl) * DD + (half << 4);
  const __bf16* w1 = Wbf + (1 * HH + h0 + nl) * DD + (half << 4);
  const __bf16* w2 = Wbf + (2 * HH + h0 + nl) * DD + (half << 4);
  const __bf16* w3 = Wbf + (3 * HH + h0 + nl) * DD + (half << 4);

  v8f a0 = {}, a1 = {}, a2 = {}, a3 = {};
  #pragma unroll
  for (int k = 0; k < DD; k += 32) {
    v16bf a = load_frag_a(arow + k);
    a0 = wmma_bf16(a, load_frag_b(w0 + k), a0);
    a1 = wmma_bf16(a, load_frag_b(w1 + k), a1);
    a2 = wmma_bf16(a, load_frag_b(w2 + k), a2);
    a3 = wmma_bf16(a, load_frag_b(w3 + k), a3);
  }

  const int h = h0 + nl;
  const float c0 = bi[h], c1 = bf_[h], c2 = bo[h], c3 = bg[h];
  #pragma unroll
  for (int v = 0; v < 8; ++v) {
    int m = m0 + (half << 3) + v;
    int b = m >> 9;            // / T
    int t = m & (TT - 1);      // % T   (tiles never straddle b since T%16==0)
    int base = ((t << 2) * BB + b) * HH + h;
    gates[base + 0 * BB * HH] = a0[v] + c0;
    gates[base + 1 * BB * HH] = a1[v] + c1;
    gates[base + 2 * BB * HH] = a2[v] + c2;
    gates[base + 3 * BB * HH] = a3[v] + c3;
  }
}

// --- phase 2: persistent recurrent scan ----------------------------------
// 32 blocks x 128 threads = 128 waves. All 4 waves of a block share one
// 16-wide h-tile -> U rows for all 4 gates (64 KB) staged in LDS once.
// Per step: hoisted gate loads, 64 LDS-fed WMMA with A prefetch, gate math,
// outputs, device-wide barrier.

__global__ __launch_bounds__(128) void lstm_scan_kernel(
    const float* __restrict__ gates, const __bf16* __restrict__ Ubf,
    __bf16* __restrict__ hbuf, unsigned* __restrict__ syn,
    const float* __restrict__ Ui_w, const float* __restrict__ Uf_w,
    const float* __restrict__ Ug_w, const float* __restrict__ out_w,
    float* __restrict__ ys, float* __restrict__ hs, float* __restrict__ fs,
    float* __restrict__ es, float* __restrict__ cds)
{
  extern __shared__ __bf16 Ulds[];   // [4][16][ULDS_STRIDE]

  const int lane = threadIdx.x & 31;
  const int wid  = blockIdx.x * (blockDim.x >> 5) + (threadIdx.x >> 5); // 0..127
  const int b0 = (wid & 3) << 4;        // batch tile (4 waves of a block)
  const int h0 = blockIdx.x << 4;       // hidden tile (shared by the block)
  const int nl = lane & 15;
  const int half = lane >> 4;
  const int h = h0 + nl;
  const int brow = b0 + (half << 3);

  // ---- stage U rows (4 gates x 16 rows x 512 K, bf16) into LDS ----
  for (int e8 = threadIdx.x; e8 < (4 * 16 * HH) / 8; e8 += blockDim.x) {
    int e = e8 << 3;
    int g = e >> 13;              // / (16*512)
    int rem = e & 8191;
    int r = rem >> 9;
    int k = rem & 511;
    *(uint4*)(Ulds + (g * 16 + r) * ULDS_STRIDE + k) =
        *(const uint4*)(Ubf + (g * HH + h0 + r) * HH + k);
  }
  __syncthreads();

  const float uid = Ui_w[h * HH + h];
  const float ufd = Uf_w[h * HH + h];
  const float ugd = Ug_w[h * HH + h];
  const float ow  = out_w[h];

  const __bf16* u0 = Ulds + (0 * 16 + nl) * ULDS_STRIDE + (half << 4);
  const __bf16* u1 = Ulds + (1 * 16 + nl) * ULDS_STRIDE + (half << 4);
  const __bf16* u2 = Ulds + (2 * 16 + nl) * ULDS_STRIDE + (half << 4);
  const __bf16* u3 = Ulds + (3 * 16 + nl) * ULDS_STRIDE + (half << 4);

  float c[8];
  #pragma unroll
  for (int v = 0; v < 8; ++v) c[v] = 0.0f;

  for (int t = 0; t < TT; ++t) {
    const __bf16* hprev = hbuf + (t & 1) * (BB * HH);
    __bf16* hnext = hbuf + ((t & 1) ^ 1) * (BB * HH);

    // hoist the (h-independent) gate pre-activation loads: overlap the GEMM
    const float* gp = gates + (t << 2) * BB * HH + h;
    float gi[8], gf[8], go[8], gg[8];
    #pragma unroll
    for (int v = 0; v < 8; ++v) {
      int b = brow + v;
      gi[v] = gp[(0 * BB + b) * HH];
      gf[v] = gp[(1 * BB + b) * HH];
      go[v] = gp[(2 * BB + b) * HH];
      gg[v] = gp[(3 * BB + b) * HH];
    }
    // warm L2 for next step's gate lines (they stream from HBM)
    if (t + 1 < TT) {
      const float* np = gates + ((t + 1) << 2) * BB * HH + brow * HH + h;
      __builtin_prefetch(np + 0 * BB * HH, 0, 1);
      __builtin_prefetch(np + 1 * BB * HH, 0, 1);
      __builtin_prefetch(np + 2 * BB * HH, 0, 1);
      __builtin_prefetch(np + 3 * BB * HH, 0, 1);
    }

    // 4 gate GEMMs: A from global h-state (prefetched 1 iter ahead),
    // B from LDS-resident U tiles
    v8f ai = {}, af = {}, ao = {}, ag = {};
    const __bf16* arow = hprev + (b0 + nl) * HH + (half << 4);
    v16bf a_cur = load_frag_a(arow);
    #pragma unroll
    for (int k = 0; k < HH; k += 32) {
      v16bf a_next = a_cur;
      if (k + 32 < HH) a_next = load_frag_a(arow + k + 32);
      ai = wmma_bf16(a_cur, load_frag_b(u0 + k), ai);
      af = wmma_bf16(a_cur, load_frag_b(u1 + k), af);
      ao = wmma_bf16(a_cur, load_frag_b(u2 + k), ao);
      ag = wmma_bf16(a_cur, load_frag_b(u3 + k), ag);
      a_cur = a_next;
    }

    float yv[8];
    #pragma unroll
    for (int v = 0; v < 8; ++v) {
      int b = brow + v;
      float iv = fsigmoid(ai[v] + gi[v]);
      float fv = fsigmoid(af[v] + gf[v]);
      float ov = fsigmoid(ao[v] + go[v]);
      float gv = ftanh(ag[v] + gg[v]);
      float cp = c[v];
      float cn = fv * cp + iv * gv;
      float tc = ftanh(cn);
      float hn = ov * tc;
      c[v] = cn;
      hnext[b * HH + h] = f2bf(hn);
      int idx = (b * TT + t) * HH + h;
      hs[idx]  = hn;
      fs[idx]  = fv;
      es[idx]  = ov * (1.0f - tc * tc);
      cds[idx] = cp * (fv * (1.0f - fv)) * ufd
               + iv * (1.0f - gv * gv) * ugd
               + gv * (iv * (1.0f - iv)) * uid;
      yv[v] = hn * ow;
    }
    // y reduction: sum the 16 lanes (h columns) of each half, one atomic per b
    #pragma unroll
    for (int v = 0; v < 8; ++v) {
      float s = yv[v];
      s += __shfl_xor(s, 8, 16);
      s += __shfl_xor(s, 4, 16);
      s += __shfl_xor(s, 2, 16);
      s += __shfl_xor(s, 1, 16);
      if (nl == 0) atomicAdd(&ys[(brow + v) * TT + t], s);
    }

    // device-wide sense-reversing barrier (once per timestep)
    __threadfence();
    __syncthreads();
    if (threadIdx.x == 0) {
      unsigned g = __hip_atomic_load(&syn[1], __ATOMIC_RELAXED, __HIP_MEMORY_SCOPE_AGENT);
      unsigned arrived = __hip_atomic_fetch_add(&syn[0], 1u, __ATOMIC_ACQ_REL,
                                                __HIP_MEMORY_SCOPE_AGENT);
      if (arrived == gridDim.x - 1) {
        __hip_atomic_store(&syn[0], 0u, __ATOMIC_RELAXED, __HIP_MEMORY_SCOPE_AGENT);
        __hip_atomic_fetch_add(&syn[1], 1u, __ATOMIC_RELEASE, __HIP_MEMORY_SCOPE_AGENT);
      } else {
        while (__hip_atomic_load(&syn[1], __ATOMIC_ACQUIRE,
                                 __HIP_MEMORY_SCOPE_AGENT) == g) {
          __builtin_amdgcn_s_sleep(1);
        }
      }
    }
    __syncthreads();
  }
}

// --- host ----------------------------------------------------------------

extern "C" void kernel_launch(void* const* d_in, const int* in_sizes, int n_in,
                              void* d_out, int out_size, void* d_ws, size_t ws_size,
                              hipStream_t stream) {
  (void)in_sizes; (void)n_in; (void)out_size; (void)ws_size;

  const float* x    = (const float*)d_in[0];
  const float* Wi_w = (const float*)d_in[1];  const float* Wi_b = (const float*)d_in[2];
  const float* Ui_w = (const float*)d_in[3];
  const float* Wf_w = (const float*)d_in[4];  const float* Wf_b = (const float*)d_in[5];
  const float* Uf_w = (const float*)d_in[6];
  const float* Wo_w = (const float*)d_in[7];  const float* Wo_b = (const float*)d_in[8];
  const float* Uo_w = (const float*)d_in[9];
  const float* Wg_w = (const float*)d_in[10]; const float* Wg_b = (const float*)d_in[11];
  const float* Ug_w = (const float*)d_in[12];
  const float* out_w = (const float*)d_in[13];
  const float* out_b = (const float*)d_in[14];

  // workspace layout (bytes)
  char* ws = (char*)d_ws;
  __bf16*   xbf   = (__bf16*)(ws + 0);                       // 8,388,608
  __bf16*   Wbf   = (__bf16*)(ws + 8388608);                 //   524,288
  __bf16*   Ubf   = (__bf16*)(ws + 8912896);                 // 2,097,152
  __bf16*   hbuf  = (__bf16*)(ws + 11010048);                //   131,072
  unsigned* syn   = (unsigned*)(ws + 11141120);              //       256
  float*    gates = (float*)(ws + 11141376);                 // 268,435,456

  float* ys  = (float*)d_out;                // [B,T,1]
  float* hs  = ys + BB * TT;                 // [B,T,H]
  float* fs  = hs + (size_t)BB * TT * HH;
  float* es  = fs + (size_t)BB * TT * HH;
  float* cds = es + (size_t)BB * TT * HH;

  // bf16 conversions
  cvt_bf16_kernel<<<(BB * TT * DD + 255) / 256, 256, 0, stream>>>(xbf, x, BB * TT * DD);
  cvt_bf16_kernel<<<(HH * DD + 255) / 256, 256, 0, stream>>>(Wbf + 0 * HH * DD, Wi_w, HH * DD);
  cvt_bf16_kernel<<<(HH * DD + 255) / 256, 256, 0, stream>>>(Wbf + 1 * HH * DD, Wf_w, HH * DD);
  cvt_bf16_kernel<<<(HH * DD + 255) / 256, 256, 0, stream>>>(Wbf + 2 * HH * DD, Wo_w, HH * DD);
  cvt_bf16_kernel<<<(HH * DD + 255) / 256, 256, 0, stream>>>(Wbf + 3 * HH * DD, Wg_w, HH * DD);
  cvt_bf16_kernel<<<(HH * HH + 255) / 256, 256, 0, stream>>>(Ubf + 0 * HH * HH, Ui_w, HH * HH);
  cvt_bf16_kernel<<<(HH * HH + 255) / 256, 256, 0, stream>>>(Ubf + 1 * HH * HH, Uf_w, HH * HH);
  cvt_bf16_kernel<<<(HH * HH + 255) / 256, 256, 0, stream>>>(Ubf + 2 * HH * HH, Uo_w, HH * HH);
  cvt_bf16_kernel<<<(HH * HH + 255) / 256, 256, 0, stream>>>(Ubf + 3 * HH * HH, Ug_w, HH * HH);

  // zero h state, reset barrier, seed ys with out_b
  init_kernel<<<256, 256, 0, stream>>>(hbuf, syn, ys, out_b);

  // phase 1: 65536 waves, 8 per block, one 16x16 tile x 4 gates per wave
  gate_gemm_kernel<<<8192, 256, 0, stream>>>(xbf, Wbf, Wi_b, Wf_b, Wo_b, Wg_b, gates);

  // phase 2: persistent scan (32 blocks x 4 waves), 66,560 B dynamic LDS for U
  size_t ulds_bytes = (size_t)4 * 16 * ULDS_STRIDE * sizeof(__bf16);
  lstm_scan_kernel<<<32, 128, ulds_bytes, stream>>>(gates, Ubf, hbuf, syn,
                                                    Ui_w, Uf_w, Ug_w, out_w,
                                                    ys, hs, fs, es, cds);
}
